// RNNDetector_64510408786169
// MI455X (gfx1250) — matrix operand order repaired and checked by reading, not compiled, not run on hardware
//
#include <hip/hip_runtime.h>
#include <hip/hip_bf16.h>
#include <stdint.h>

// ---------------------------------------------------------------------------
// 2-layer LSTM (H=256) + FC head, fused persistent kernel for MI455X (gfx1250)
//
//  * Batch (128) split across 8 workgroups (16 batches each); recurrence is
//    workgroup-local: 3 s_barriers per timestep, no grid sync.
//  * 512 threads = 16 waves (4/SIMD32); wave w owns hidden units [16w,16w+16)
//    => 4 WMMA N-tiles (one per gate i,f,g,o). Cell state c stays in VGPRs for
//    all 2048 steps; h0/h1 live in LDS as ping-pong f16 [2][16 x 256].
//  * v_wmma_f32_16x16x32_f16; weights pre-packed f16 fragment-major (1KB per
//    32x16 B-fragment), streamed from L2 every step. A zero-SGPR launder stops
//    the compiler hoisting the t-invariant weight loads (scratch spill) while
//    keeping pointer provenance so loads lower to global_load_b128 (not flat).
//  * Branch-free activations: v_rcp/v_exp2 sigmoid, v_tanh_f32 when available.
//  * Layer-1 input projection fused into its recurrent GEMM; FC head fused.
// ---------------------------------------------------------------------------

typedef __attribute__((ext_vector_type(16))) _Float16 v16h;
typedef __attribute__((ext_vector_type(8)))  float    v8f;

#define T_LEN 2048
#define HID   256
#define G4    1024
#define PADV  (-100.0f)

#define LOG2E  1.44269504088896f

__device__ __forceinline__ float fast_sigmoid(float x) {
  // 1/(1+e^-x) ; v_exp_f32 is exp2 -> scale by log2(e). Saturates cleanly:
  // exp2(+inf)=inf -> rcp(inf)=0 ; exp2(-inf)=0 -> rcp(1)=1.
  return __builtin_amdgcn_rcpf(1.0f + __builtin_amdgcn_exp2f(x * -LOG2E));
}

__device__ __forceinline__ float fast_tanh(float x) {
#if __has_builtin(__builtin_amdgcn_tanhf)
  return __builtin_amdgcn_tanhf(x);
#else
  return 2.0f * __builtin_amdgcn_rcpf(1.0f + __builtin_amdgcn_exp2f(x * (-2.0f * LOG2E))) - 1.0f;
#endif
}

// B-fragment (32x16, f16): fragment (nt,kt) is 1KB contiguous, lane reads 32B.
// lanes 0-15 -> N = lane, K = 0..7,16..23 ; lanes 16-31 -> N = lane-16, K = 8..15,24..31.
__device__ __forceinline__ v16h load_b_frag(const _Float16* __restrict__ pw,
                                            int nt, int kt, int lane) {
  const uint4* q = (const uint4*)(pw + ((nt * 8 + kt) << 9)) + (lane << 1);
  union { v16h v; uint4 u[2]; } t;
  t.u[0] = q[0];
  t.u[1] = q[1];
  return t.v;
}

// A-fragment (16x32, f16) from LDS h buffer [16][256] row-major.
// lane: M = lane&15 ; K = kc*32 + (lane>>4)*8 + {0..7, 16..23}.
__device__ __forceinline__ v16h load_a_frag(const _Float16* buf, int kc, int lane) {
  const int m  = lane & 15;
  const int k0 = kc * 32 + (lane >> 4) * 8;
  const uint4* q = (const uint4*)(buf + m * HID + k0);
  union { v16h v; uint4 u[2]; } t;
  t.u[0] = q[0];   // K = k0 .. k0+7
  t.u[1] = q[2];   // K = k0+16 .. k0+23  (+32 bytes)
  return t.v;
}

// ---------------------------------------------------------------------------
// Pack kernel: W ([1024 x 256] f32, row n = gate output, col k) -> f16
// fragment-major B layout; plus bias sums b_ih + b_hh.
// ---------------------------------------------------------------------------
__global__ void lstm_pack_kernel(const float* __restrict__ Whh0,
                                 const float* __restrict__ Wih1,
                                 const float* __restrict__ Whh1,
                                 const float* __restrict__ bih0,
                                 const float* __restrict__ bhh0,
                                 const float* __restrict__ bih1,
                                 const float* __restrict__ bhh1,
                                 _Float16* __restrict__ pWhh0,
                                 _Float16* __restrict__ pWih1,
                                 _Float16* __restrict__ pWhh1,
                                 float* __restrict__ b0sum,
                                 float* __restrict__ b1sum) {
  const int per = G4 * HID;              // 262144 halves per matrix
  const int total = 3 * per + 2 * G4;
  for (int i = blockIdx.x * blockDim.x + threadIdx.x; i < total;
       i += gridDim.x * blockDim.x) {
    if (i < 3 * per) {
      const int mi = i / per;
      const int r  = i % per;
      const int frag = r >> 9;
      const int lane = (r >> 4) & 31;
      const int e    = r & 15;
      const int nt = frag >> 3;
      const int kt = frag & 7;
      const int n = nt * 16 + (lane & 15);
      const int k = kt * 32 + (lane >> 4) * 8 + (e & 7) + (e >> 3) * 16;
      const float* W = (mi == 0) ? Whh0 : ((mi == 1) ? Wih1 : Whh1);
      _Float16*    P = (mi == 0) ? pWhh0 : ((mi == 1) ? pWih1 : pWhh1);
      P[r] = (_Float16)W[n * HID + k];
    } else {
      const int j = i - 3 * per;
      if (j < G4) b0sum[j] = bih0[j] + bhh0[j];
      else        b1sum[j - G4] = bih1[j - G4] + bhh1[j - G4];
    }
  }
}

// ---------------------------------------------------------------------------
// Fused persistent LSTM kernel. grid = 8 WGs x 512 threads (16 waves).
// Wave w owns hidden units [16w, 16w+16): N-tiles { 16g + w : g=0..3 }.
// C/D f32 fragment layout: VGPR r, lane l: M = r + 8*(l>=16), N = l&15.
// ---------------------------------------------------------------------------
__launch_bounds__(512, 1)
__global__ void lstm2_fused_kernel(const float* __restrict__ y,
                                   const _Float16* __restrict__ pWhh0_in,
                                   const _Float16* __restrict__ pWih1_in,
                                   const _Float16* __restrict__ pWhh1_in,
                                   const float* __restrict__ b0sum,
                                   const float* __restrict__ b1sum,
                                   const float* __restrict__ Wih0,  // [1024,4]
                                   const float* __restrict__ Wfc,   // [2,256]
                                   const float* __restrict__ bfc,   // [2]
                                   float* __restrict__ out) {       // [128,2048,2]
  __shared__ _Float16 h0buf[2][16 * HID];   // ping-pong layer-0 hidden (16 KB)
  __shared__ _Float16 h1buf[2][16 * HID];   // ping-pong layer-1 hidden (16 KB)
  __shared__ __align__(16) float wih0[G4 * 4];  // 16 KB layer-0 input weights
  __shared__ float b0l[G4];
  __shared__ float b1l[G4];
  __shared__ float wfcl[2 * HID];
  __shared__ __align__(16) float wbufs[16][4];  // window taps per step
  __shared__ float bfcl[2];

  const int tid  = threadIdx.x;
  const int wave = tid >> 5;             // 0..15
  const int lane = tid & 31;
  const int lrow = lane & 15;
  const int hi   = lane >> 4;
  const int mb   = blockIdx.x * 16;      // batch base for this WG

  for (int i = tid; i < G4; i += 512)
    ((float4*)wih0)[i] = ((const float4*)Wih0)[i];
  for (int i = tid; i < G4; i += 512) { b0l[i] = b0sum[i]; b1l[i] = b1sum[i]; }
  for (int i = tid; i < 2 * HID; i += 512) wfcl[i] = Wfc[i];
  if (tid < 2) bfcl[tid] = bfc[tid];
  for (int i = tid; i < 16 * HID; i += 512) {
    h0buf[0][i] = (_Float16)0.0f; h0buf[1][i] = (_Float16)0.0f;
    h1buf[0][i] = (_Float16)0.0f; h1buf[1][i] = (_Float16)0.0f;
  }

  float c0s[8], c1s[8];
#pragma unroll
  for (int r = 0; r < 8; ++r) { c0s[r] = 0.0f; c1s[r] = 0.0f; }

  __syncthreads();

  for (int t = 0; t < T_LEN; ++t) {
    const int rp = t & 1;        // read parity
    const int wp = rp ^ 1;       // write parity

    // Launder a zero SGPR offset: keeps pointer provenance (global_load) but
    // stops the compiler proving the weight loads t-invariant (no hoist/spill).
    uint32_t lz = 0;
    asm volatile("" : "+s"(lz));
    const _Float16* __restrict__ pWhh0 = pWhh0_in + lz;
    const _Float16* __restrict__ pWih1 = pWih1_in + lz;
    const _Float16* __restrict__ pWhh1 = pWhh1_in + lz;

    // ---- stage window taps y[t-3..t] for the 16 batches ----
    if (tid < 64) {
      const int b = tid >> 2, d = tid & 3;
      const int ti = t - 3 + d;
      wbufs[b][d] = (ti >= 0) ? y[(mb + b) * T_LEN + ti] : PADV;
    }
    __syncthreads();  // B1: taps ready (also orders vs prev-step writes)

    v8f acc[4];

    // ================= layer 0 =================
    // init accumulators with bias + 4-tap input projection (float4 LDS loads)
    float4 taps[8];
#pragma unroll
    for (int r = 0; r < 8; ++r) taps[r] = *(const float4*)wbufs[r + 8 * hi];
#pragma unroll
    for (int g = 0; g < 4; ++g) {
      const int n = 256 * g + 16 * wave + lrow;
      const float4 w = *(const float4*)(&wih0[n * 4]);
      const float base = b0l[n];
#pragma unroll
      for (int r = 0; r < 8; ++r) {
        acc[g][r] = base + taps[r].x * w.x + taps[r].y * w.y +
                           taps[r].z * w.z + taps[r].w * w.w;
      }
    }

    // recurrent GEMM: gates += h0(t-1) @ Whh0^T   (K = 256)
#pragma unroll 2
    for (int kc = 0; kc < 8; ++kc) {
      const v16h a = load_a_frag(h0buf[rp], kc, lane);
      v16h bf[4];
#pragma unroll
      for (int g = 0; g < 4; ++g)
        bf[g] = load_b_frag(pWhh0, 16 * g + wave, kc, lane);
#pragma unroll
      for (int g = 0; g < 4; ++g)
        acc[g] = __builtin_amdgcn_wmma_f32_16x16x32_f16(
            false, a, false, bf[g], (short)0, acc[g], false, false);
    }

#pragma unroll
    for (int r = 0; r < 8; ++r) {
      const float iv = fast_sigmoid(acc[0][r]);
      const float fv = fast_sigmoid(acc[1][r]);
      const float gv = fast_tanh(acc[2][r]);
      const float ov = fast_sigmoid(acc[3][r]);
      const float c = fv * c0s[r] + iv * gv;
      c0s[r] = c;
      const float h = ov * fast_tanh(c);
      h0buf[wp][(r + 8 * hi) * HID + 16 * wave + lrow] = (_Float16)h;
    }
    __syncthreads();  // B2: h0(t) complete in LDS

    // ================= layer 1 =================
#pragma unroll
    for (int g = 0; g < 4; ++g) {
      const float base = b1l[256 * g + 16 * wave + lrow];
#pragma unroll
      for (int r = 0; r < 8; ++r) acc[g][r] = base;
    }

    // input projection: gates += h0(t) @ Wih1^T   (K = 256)
#pragma unroll 2
    for (int kc = 0; kc < 8; ++kc) {
      const v16h a = load_a_frag(h0buf[wp], kc, lane);
      v16h bf[4];
#pragma unroll
      for (int g = 0; g < 4; ++g)
        bf[g] = load_b_frag(pWih1, 16 * g + wave, kc, lane);
#pragma unroll
      for (int g = 0; g < 4; ++g)
        acc[g] = __builtin_amdgcn_wmma_f32_16x16x32_f16(
            false, a, false, bf[g], (short)0, acc[g], false, false);
    }
    // recurrent part: gates += h1(t-1) @ Whh1^T   (K = 256)
#pragma unroll 2
    for (int kc = 0; kc < 8; ++kc) {
      const v16h a = load_a_frag(h1buf[rp], kc, lane);
      v16h bf[4];
#pragma unroll
      for (int g = 0; g < 4; ++g)
        bf[g] = load_b_frag(pWhh1, 16 * g + wave, kc, lane);
#pragma unroll
      for (int g = 0; g < 4; ++g)
        acc[g] = __builtin_amdgcn_wmma_f32_16x16x32_f16(
            false, a, false, bf[g], (short)0, acc[g], false, false);
    }

#pragma unroll
    for (int r = 0; r < 8; ++r) {
      const float iv = fast_sigmoid(acc[0][r]);
      const float fv = fast_sigmoid(acc[1][r]);
      const float gv = fast_tanh(acc[2][r]);
      const float ov = fast_sigmoid(acc[3][r]);
      const float c = fv * c1s[r] + iv * gv;
      c1s[r] = c;
      const float h = ov * fast_tanh(c);
      h1buf[wp][(r + 8 * hi) * HID + 16 * wave + lrow] = (_Float16)h;
    }
    __syncthreads();  // B3: h1(t) complete in LDS

    // ================= FC head =================
    // wave w handles batch b = w; lane halves handle class 0/1.
    {
      const int cls = hi;                 // class 0 or 1
      const int b   = wave;
      float s = 0.0f;
#pragma unroll
      for (int jj = 0; jj < 16; ++jj) {
        const int j = lrow * 16 + jj;
        s += (float)h1buf[wp][b * HID + j] * wfcl[cls * HID + j];
      }
      s += __shfl_xor(s, 1);
      s += __shfl_xor(s, 2);
      s += __shfl_xor(s, 4);
      s += __shfl_xor(s, 8);
      if (lrow == 0)
        out[((size_t)(mb + b) * T_LEN + t) * 2 + cls] = s + bfcl[cls];
    }
    // ping-pong: no read-barrier needed before next step's writes
  }
}

// ---------------------------------------------------------------------------
// Host-side launch.  Inputs (all f32, setup_inputs order):
//  0:y 1:W_ih0 2:W_hh0 3:b_ih0 4:b_hh0 5:W_ih1 6:W_hh1 7:b_ih1 8:b_hh1
//  9:W_fc 10:b_fc
// Workspace: 3 x 512KB packed f16 weights + 8KB bias sums (~1.6 MB).
// ---------------------------------------------------------------------------
extern "C" void kernel_launch(void* const* d_in, const int* in_sizes, int n_in,
                              void* d_out, int out_size, void* d_ws, size_t ws_size,
                              hipStream_t stream) {
  (void)in_sizes; (void)n_in; (void)out_size; (void)ws_size;
  const float* y    = (const float*)d_in[0];
  const float* Wih0 = (const float*)d_in[1];
  const float* Whh0 = (const float*)d_in[2];
  const float* bih0 = (const float*)d_in[3];
  const float* bhh0 = (const float*)d_in[4];
  const float* Wih1 = (const float*)d_in[5];
  const float* Whh1 = (const float*)d_in[6];
  const float* bih1 = (const float*)d_in[7];
  const float* bhh1 = (const float*)d_in[8];
  const float* Wfc  = (const float*)d_in[9];
  const float* bfc  = (const float*)d_in[10];

  char* ws = (char*)d_ws;
  const size_t MATB = (size_t)G4 * HID * sizeof(_Float16);  // 512 KB
  _Float16* pWhh0 = (_Float16*)(ws);
  _Float16* pWih1 = (_Float16*)(ws + MATB);
  _Float16* pWhh1 = (_Float16*)(ws + 2 * MATB);
  float*    b0sum = (float*)(ws + 3 * MATB);
  float*    b1sum = b0sum + G4;

  lstm_pack_kernel<<<512, 256, 0, stream>>>(Whh0, Wih1, Whh1,
                                            bih0, bhh0, bih1, bhh1,
                                            pWhh0, pWih1, pWhh1, b0sum, b1sum);

  lstm2_fused_kernel<<<8, 512, 0, stream>>>(y, pWhh0, pWih1, pWhh1,
                                            b0sum, b1sum, Wih0, Wfc, bfc,
                                            (float*)d_out);
}
